// IncrementalRobertaSelfAttentionCross_46359876993470
// MI455X (gfx1250) — matrix operand order repaired and checked by reading, not compiled, not run on hardware
//
#include <hip/hip_runtime.h>

typedef __attribute__((ext_vector_type(16))) __bf16 v16bf;
typedef __attribute__((ext_vector_type(8)))  float  v8f;

#define B_   4
#define H_   16
#define D_   1024
#define HD_  64
#define SQ_  1024
#define SKV_ 2048
#define LOG2E 1.44269504088896340736f

__device__ __forceinline__ unsigned short f2bf(float x) {
  union { float f; unsigned u; } a; a.f = x;
  unsigned r = a.u + 0x7FFFu + ((a.u >> 16) & 1u);   // round-to-nearest-even
  return (unsigned short)(r >> 16);
}
__device__ __forceinline__ unsigned pack2(float a, float b) {
  return (unsigned)f2bf(a) | ((unsigned)f2bf(b) << 16);
}

union FragU { uint4 q[2]; v16bf v; };

// A-fragment (16x32 bf16, M = lane%16 row of a row-major matrix):
// lane chunks at K = {hi*8 .. hi*8+7} and {16+hi*8 .. 16+hi*8+7}
__device__ __forceinline__ v16bf load_afrag(const unsigned short* rowp, int hi) {
  FragU u;
  u.q[0] = *reinterpret_cast<const uint4*>(rowp + (hi << 3));
  u.q[1] = *reinterpret_cast<const uint4*>(rowp + (hi << 3) + 16);
  return u.v;
}
// B-fragment (32x16 bf16) loaded from row-major B^T (N = lane%16 row):
// lane chunks at K = {hi*8 .. hi*8+7} and {16+hi*8 .. 16+hi*8+7}
__device__ __forceinline__ v16bf load_bfrag(const unsigned short* rowp, int hi) {
  const uint4* q = reinterpret_cast<const uint4*>(rowp + (hi << 3));
  FragU u; u.q[0] = q[0]; u.q[1] = q[2];
  return u.v;
}
__device__ __forceinline__ v8f wmma_bf16(v16bf a, v16bf b, v8f c) {
  return __builtin_amdgcn_wmma_f32_16x16x32_bf16(false, a, false, b, (short)0, c,
                                                 false, false);
}

// ---------------------------------------------------------------------------
// Kernel 1: pack concat(encoder, decoder) -> bf16 Xall [B][2048][1024]
// ---------------------------------------------------------------------------
__global__ __launch_bounds__(256)
void cvt_inputs_kernel(const float* __restrict__ hid, const float* __restrict__ enc,
                       unsigned short* __restrict__ xall) {
  long i = (long)blockIdx.x * blockDim.x + threadIdx.x;   // group of 4 elems
  const long N4 = (long)B_ * SKV_ * D_ / 4;
  if (i >= N4) return;
  long e = i * 4;
  int b = (int)(e / ((long)SKV_ * D_));
  long r = e - (long)b * SKV_ * D_;
  int s = (int)(r / D_);
  int k = (int)(r % D_);
  float4 src;
  if (s < SQ_) src = *reinterpret_cast<const float4*>(enc + ((long)b * SQ_ + s) * D_ + k);
  else         src = *reinterpret_cast<const float4*>(hid + ((long)b * SQ_ + (s - SQ_)) * D_ + k);
  uint2 pk;
  pk.x = pack2(src.x, src.y);
  pk.y = pack2(src.z, src.w);
  *reinterpret_cast<uint2*>(xall + e) = pk;
}

// ---------------------------------------------------------------------------
// Kernel 2: Wt[n][k] = bf16(W[k][n])   (LDS-tiled transpose, 32x32 tiles)
// ---------------------------------------------------------------------------
__global__ __launch_bounds__(256)
void wtrans_kernel(const float* __restrict__ W, unsigned short* __restrict__ Wt) {
  __shared__ unsigned short tile[32][33];
  int k0 = blockIdx.x * 32, n0 = blockIdx.y * 32;
  int tx = threadIdx.x;        // 0..31
  int ty = threadIdx.y;        // 0..7
  #pragma unroll
  for (int i = 0; i < 32; i += 8)
    tile[ty + i][tx] = f2bf(W[(long)(k0 + ty + i) * D_ + (n0 + tx)]);
  __syncthreads();
  #pragma unroll
  for (int i = 0; i < 32; i += 8)
    Wt[(long)(n0 + ty + i) * D_ + (k0 + tx)] = tile[tx][ty + i];
}

// ---------------------------------------------------------------------------
// Kernel 3: projection Y = X * W + b via bf16 WMMA, fp32 accumulate.
// One wave -> 32 rows x 64 cols (8 WMMAs / K-step), double-buffered loads.
// mode: 0=Q, 1=K, 2=V(transposed out)
// ---------------------------------------------------------------------------
__global__ __launch_bounds__(128)
void proj_kernel(const unsigned short* __restrict__ X,
                 const unsigned short* __restrict__ Wt,
                 const float* __restrict__ bias,
                 unsigned short* __restrict__ out, int mode) {
  int lane = threadIdx.x & 31;
  int nlo = lane & 15, hi = lane >> 4;
  int w = blockIdx.x * 4 + (threadIdx.x >> 5);
  int m0 = (w >> 4) << 5;           // 32-row tile base
  int n0 = (w & 15) << 6;           // 64-wide col group (one head)

  int mr0 = m0 + nlo;               // A rows this lane loads (two sub-tiles)
  int mr1 = m0 + 16 + nlo;
  long xr0 = (mode == 0) ? ((long)(mr0 >> 10) * SKV_ + SQ_ + (mr0 & (SQ_ - 1))) : (long)mr0;
  long xr1 = (mode == 0) ? ((long)(mr1 >> 10) * SKV_ + SQ_ + (mr1 & (SQ_ - 1))) : (long)mr1;
  const unsigned short* xp0 = X + xr0 * D_;
  const unsigned short* xp1 = X + xr1 * D_;
  const unsigned short* wp0 = Wt + (long)(n0 +  0 + nlo) * D_;
  const unsigned short* wp1 = Wt + (long)(n0 + 16 + nlo) * D_;
  const unsigned short* wp2 = Wt + (long)(n0 + 32 + nlo) * D_;
  const unsigned short* wp3 = Wt + (long)(n0 + 48 + nlo) * D_;

  v8f acc[2][4];
  #pragma unroll
  for (int s = 0; s < 2; s++)
    #pragma unroll
    for (int t = 0; t < 4; t++) acc[s][t] = v8f{0, 0, 0, 0, 0, 0, 0, 0};

  // software pipeline: preload k=0, then each iter issues k+32 before consuming k
  v16bf ac0 = load_afrag(xp0, hi), ac1 = load_afrag(xp1, hi);
  v16bf bc0 = load_bfrag(wp0, hi), bc1 = load_bfrag(wp1, hi);
  v16bf bc2 = load_bfrag(wp2, hi), bc3 = load_bfrag(wp3, hi);

  for (int k = 32; k <= D_; k += 32) {
    int kn = (k < D_) ? k : 0;                       // wrap: keeps EXEC uniform
    v16bf an0 = load_afrag(xp0 + kn, hi);
    v16bf an1 = load_afrag(xp1 + kn, hi);
    v16bf bn0 = load_bfrag(wp0 + kn, hi);
    v16bf bn1 = load_bfrag(wp1 + kn, hi);
    v16bf bn2 = load_bfrag(wp2 + kn, hi);
    v16bf bn3 = load_bfrag(wp3 + kn, hi);

    acc[0][0] = wmma_bf16(ac0, bc0, acc[0][0]);
    acc[1][0] = wmma_bf16(ac1, bc0, acc[1][0]);
    acc[0][1] = wmma_bf16(ac0, bc1, acc[0][1]);
    acc[1][1] = wmma_bf16(ac1, bc1, acc[1][1]);
    acc[0][2] = wmma_bf16(ac0, bc2, acc[0][2]);
    acc[1][2] = wmma_bf16(ac1, bc2, acc[1][2]);
    acc[0][3] = wmma_bf16(ac0, bc3, acc[0][3]);
    acc[1][3] = wmma_bf16(ac1, bc3, acc[1][3]);

    ac0 = an0; ac1 = an1;
    bc0 = bn0; bc1 = bn1; bc2 = bn2; bc3 = bn3;
  }

  int h = n0 >> 6;
  if (mode < 2) {
    // out[((b*16+h)*S + s)*64 + hd], bf16
    int S = (mode == 0) ? SQ_ : SKV_;
    int shift = (mode == 0) ? 10 : 11;
    #pragma unroll
    for (int sub = 0; sub < 2; sub++) {
      #pragma unroll
      for (int t = 0; t < 4; t++) {
        float bb = bias[n0 + t * 16 + nlo];
        int hd = t * 16 + nlo;
        #pragma unroll
        for (int r = 0; r < 8; r++) {
          int m = m0 + sub * 16 + r + (hi << 3);
          int bidx = m >> shift;
          int s = m & (S - 1);
          out[(((long)bidx * H_ + h) * S + s) * HD_ + hd] = f2bf(acc[sub][t][r] + bb);
        }
      }
    }
  } else {
    // Vt: out[((b*16+h)*64 + hd)*2048 + s], 8 consecutive s -> one b128 store
    #pragma unroll
    for (int sub = 0; sub < 2; sub++) {
      int m = m0 + sub * 16 + (hi << 3);
      int bidx = m >> 11;
      int s = m & (SKV_ - 1);
      #pragma unroll
      for (int t = 0; t < 4; t++) {
        float bb = bias[n0 + t * 16 + nlo];
        int hd = t * 16 + nlo;
        uint4 pkt;
        pkt.x = pack2(acc[sub][t][0] + bb, acc[sub][t][1] + bb);
        pkt.y = pack2(acc[sub][t][2] + bb, acc[sub][t][3] + bb);
        pkt.z = pack2(acc[sub][t][4] + bb, acc[sub][t][5] + bb);
        pkt.w = pack2(acc[sub][t][6] + bb, acc[sub][t][7] + bb);
        *reinterpret_cast<uint4*>(out + (((long)bidx * H_ + h) * HD_ + hd) * SKV_ + s) = pkt;
      }
    }
  }
}

// ---------------------------------------------------------------------------
// Kernel 4: flash attention. One wave = one (b,h,16-query) tile, full HD=64.
// K fragments double-buffered across iterations; V fragments issued early so
// softmax + S WMMAs hide their latency.
// ---------------------------------------------------------------------------
__global__ __launch_bounds__(128)
void attn_kernel(const unsigned short* __restrict__ Q,
                 const unsigned short* __restrict__ K,
                 const unsigned short* __restrict__ Vt,
                 const float* __restrict__ mask,
                 float* __restrict__ out) {
  __shared__ alignas(16) unsigned short pbuf[4][16][40];  // wave-private P relayout
  int wid = threadIdx.x >> 5;
  int lane = threadIdx.x & 31;
  int nlo = lane & 15, hi = lane >> 4;
  int g = blockIdx.x * 4 + wid;
  int bh = g >> 6;                 // b*16+h
  int qt = g & 63;
  int b = bh >> 4, h = bh & 15;
  int q0 = qt << 4;

  const unsigned short* Qp = Q + ((long)bh * SQ_ + q0 + nlo) * HD_;
  const unsigned short* Kp = K + (long)bh * SKV_ * HD_;
  const unsigned short* Vp = Vt + (long)bh * HD_ * SKV_;
  const float* mp = mask + (long)b * SKV_;

  v16bf qa0 = load_afrag(Qp, hi);        // hd 0..31
  v16bf qa1 = load_afrag(Qp + 32, hi);   // hd 32..63

  v8f O0{0,0,0,0,0,0,0,0}, O1{0,0,0,0,0,0,0,0}, O2{0,0,0,0,0,0,0,0}, O3{0,0,0,0,0,0,0,0};
  float mrow[8], lrow[8];
  #pragma unroll
  for (int r = 0; r < 8; r++) { mrow[r] = -1.0e30f; lrow[r] = 0.0f; }

  unsigned short* pw = &pbuf[wid][0][0];
  volatile unsigned short* pv = pw;

  // preload K fragments for j = 0
  v16bf k00 = load_bfrag(Kp + (long)nlo * HD_,             hi);
  v16bf k01 = load_bfrag(Kp + (long)nlo * HD_ + 32,        hi);
  v16bf k10 = load_bfrag(Kp + (long)(16 + nlo) * HD_,      hi);
  v16bf k11 = load_bfrag(Kp + (long)(16 + nlo) * HD_ + 32, hi);

  for (int j = 0; j < SKV_; j += 32) {
    // issue V fragment loads for this chunk (consumed after softmax)
    v16bf vf0 = load_bfrag(Vp + (long)( 0 + nlo) * SKV_ + j, hi);
    v16bf vf1 = load_bfrag(Vp + (long)(16 + nlo) * SKV_ + j, hi);
    v16bf vf2 = load_bfrag(Vp + (long)(32 + nlo) * SKV_ + j, hi);
    v16bf vf3 = load_bfrag(Vp + (long)(48 + nlo) * SKV_ + j, hi);

    v8f s0{0,0,0,0,0,0,0,0}, s1{0,0,0,0,0,0,0,0};
    s0 = wmma_bf16(qa0, k00, s0);
    s0 = wmma_bf16(qa1, k01, s0);
    s1 = wmma_bf16(qa0, k10, s1);
    s1 = wmma_bf16(qa1, k11, s1);

    // issue next chunk's K fragment loads (wrap keeps addresses in-buffer)
    {
      int jn = (j + 32) & (SKV_ - 1);
      const unsigned short* kr0 = Kp + (long)(jn + nlo) * HD_;
      const unsigned short* kr1 = kr0 + 16 * HD_;
      k00 = load_bfrag(kr0,      hi);
      k01 = load_bfrag(kr0 + 32, hi);
      k10 = load_bfrag(kr1,      hi);
      k11 = load_bfrag(kr1 + 32, hi);
      __builtin_prefetch(Kp + (long)(((j + 64) & (SKV_ - 1)) + nlo) * HD_, 0, 1);
    }

    float mk0 = mp[j + nlo], mk1 = mp[j + 16 + nlo];
    #pragma unroll
    for (int r = 0; r < 8; r++) {
      float a0 = s0[r] * 0.125f + mk0;                 // 1/sqrt(64) + additive mask
      float a1 = s1[r] * 0.125f + mk1;
      float mx = fmaxf(a0, a1);
      mx = fmaxf(mx, __shfl_xor(mx, 1));
      mx = fmaxf(mx, __shfl_xor(mx, 2));
      mx = fmaxf(mx, __shfl_xor(mx, 4));
      mx = fmaxf(mx, __shfl_xor(mx, 8));               // row max within 16-lane half
      float mnew = fmaxf(mrow[r], mx);
      float alpha = exp2f((mrow[r] - mnew) * LOG2E);
      float p0 = exp2f((a0 - mnew) * LOG2E);
      float p1 = exp2f((a1 - mnew) * LOG2E);
      mrow[r] = mnew;
      float ps = p0 + p1;
      ps += __shfl_xor(ps, 1);
      ps += __shfl_xor(ps, 2);
      ps += __shfl_xor(ps, 4);
      ps += __shfl_xor(ps, 8);
      lrow[r] = lrow[r] * alpha + ps;
      O0[r] *= alpha; O1[r] *= alpha; O2[r] *= alpha; O3[r] *= alpha;
      int row = r + (hi << 3);                          // C-layout row
      pv[row * 40 + nlo]      = f2bf(p0);               // relayout C -> A via LDS
      pv[row * 40 + 16 + nlo] = f2bf(p1);
    }
    __builtin_amdgcn_wave_barrier();
    asm volatile("s_wait_dscnt 0" ::: "memory");        // wave-local LDS visibility
    v16bf pa = load_afrag(pw + nlo * 40, hi);           // P as A-fragment (16x32)
    asm volatile("" ::: "memory");
    O0 = wmma_bf16(pa, vf0, O0);
    O1 = wmma_bf16(pa, vf1, O1);
    O2 = wmma_bf16(pa, vf2, O2);
    O3 = wmma_bf16(pa, vf3, O3);
    __builtin_amdgcn_wave_barrier();
    asm volatile("" ::: "memory");                      // keep next stores after loads
  }

  // normalize by softmax denominator, write fp32 out[b][q][h*64+hd]
  #pragma unroll
  for (int r = 0; r < 8; r++) {
    float inv = 1.0f / lrow[r];
    int qq = q0 + r + (hi << 3);
    float* op = out + ((long)b * SQ_ + qq) * D_ + h * HD_ + nlo;
    op[0]  = O0[r] * inv;
    op[16] = O1[r] * inv;
    op[32] = O2[r] * inv;
    op[48] = O3[r] * inv;
  }
}

// ---------------------------------------------------------------------------
extern "C" void kernel_launch(void* const* d_in, const int* in_sizes, int n_in,
                              void* d_out, int out_size, void* d_ws, size_t ws_size,
                              hipStream_t stream) {
  const float* hid  = (const float*)d_in[0];
  const float* enc  = (const float*)d_in[1];
  const float* mask = (const float*)d_in[2];
  const float* Wq   = (const float*)d_in[3];
  const float* bq   = (const float*)d_in[4];
  const float* Wk   = (const float*)d_in[5];
  const float* bk   = (const float*)d_in[6];
  const float* Wv   = (const float*)d_in[7];
  const float* bv   = (const float*)d_in[8];
  float* out = (float*)d_out;

  unsigned short* Xall = (unsigned short*)d_ws;                 // [4][2048][1024]
  unsigned short* Wtq  = Xall + (size_t)B_ * SKV_ * D_;         // [1024][1024] each
  unsigned short* Wtk  = Wtq + (size_t)D_ * D_;
  unsigned short* Wtv  = Wtk + (size_t)D_ * D_;
  unsigned short* Qb   = Wtv + (size_t)D_ * D_;                 // [64][1024][64]
  unsigned short* Kb   = Qb + (size_t)B_ * H_ * SQ_ * HD_;      // [64][2048][64]
  unsigned short* Vtb  = Kb + (size_t)B_ * H_ * SKV_ * HD_;     // [64][64][2048]

  // 1) pack inputs to bf16
  {
    long n4 = (long)B_ * SKV_ * D_ / 4;
    cvt_inputs_kernel<<<(unsigned)((n4 + 255) / 256), 256, 0, stream>>>(hid, enc, Xall);
  }
  // 2) transpose+convert weights
  {
    dim3 g(D_ / 32, D_ / 32), blk(32, 8);
    wtrans_kernel<<<g, blk, 0, stream>>>(Wq, Wtq);
    wtrans_kernel<<<g, blk, 0, stream>>>(Wk, Wtk);
    wtrans_kernel<<<g, blk, 0, stream>>>(Wv, Wtv);
  }
  // 3) projections (Q over decoder rows, K/V over concat rows), 32-row tiles
  proj_kernel<<<(B_ * SQ_  / 32) * 16 / 4, 128, 0, stream>>>(Xall, Wtq, bq, Qb,  0);
  proj_kernel<<<(B_ * SKV_ / 32) * 16 / 4, 128, 0, stream>>>(Xall, Wtk, bk, Kb,  1);
  proj_kernel<<<(B_ * SKV_ / 32) * 16 / 4, 128, 0, stream>>>(Xall, Wtv, bv, Vtb, 2);
  // 4) flash attention: 4096 wave tiles, 4 waves/block
  attn_kernel<<<B_ * H_ * (SQ_ / 16) / 4, 128, 0, stream>>>(Qb, Kb, Vtb, mask, out);
}
